// Net_84782654423525
// MI455X (gfx1250) — matrix operand order repaired and checked by reading, not compiled, not run on hardware
//
#include <hip/hip_runtime.h>

#define N_NODES 20000
#define FDIM    128
#define EDGES   640000
#define WIN     6
#define UNITS   256
#define GNCOLS  1024      // 4 * UNITS
#define BN_EPS  1e-3f

typedef __attribute__((ext_vector_type(16))) __bf16 v16bf;
typedef __attribute__((ext_vector_type(8)))  __bf16 v8bf;
typedef __attribute__((ext_vector_type(8)))  float  v8f;

// ---------------------------------------------------------------------------
// Graph phase
// ---------------------------------------------------------------------------

// 32 lanes per edge, 4 floats per lane (F=128). Works for both scatters:
// pass cnt!=null for the mean-aggregate pass.
__global__ void scatter_kernel(const float* __restrict__ X,
                               const int* __restrict__ src,
                               const int* __restrict__ dst,
                               float* __restrict__ acc,
                               float* __restrict__ cnt,
                               unsigned total) {
  unsigned gid = blockIdx.x * blockDim.x + threadIdx.x;
  if (gid >= total) return;
  unsigned e = gid >> 5;                 // global edge id over [W, E]
  int f = (int)(gid & 31u) * 4;
  int t = (int)(e / EDGES);
  int s = src[e];
  int d = dst[e];
  const float4 v = *(const float4*)&X[((size_t)t * N_NODES + s) * FDIM + f];
  float* a = &acc[((size_t)t * N_NODES + d) * FDIM + f];
  atomicAdd(a + 0, v.x);
  atomicAdd(a + 1, v.y);
  atomicAdd(a + 2, v.z);
  atomicAdd(a + 3, v.w);
  if (cnt != nullptr && (gid & 31u) == 0u)
    atomicAdd(&cnt[(size_t)t * N_NODES + d], 1.0f);
}

// mean -> relu -> BN; writes h1 (f32, for scatter #2), bf16 into seq[:, t, 0:F],
// and re-zeroes the accumulator so the same buffer serves scatter #2.
__global__ void mpnn1_post(float* __restrict__ s1,
                           const float* __restrict__ cnt,
                           const float* __restrict__ g, const float* __restrict__ b,
                           const float* __restrict__ m, const float* __restrict__ v,
                           float* __restrict__ h1f, __bf16* __restrict__ seq) {
  unsigned gid = blockIdx.x * blockDim.x + threadIdx.x;   // W*N*F threads
  int t = (int)(gid / (N_NODES * FDIM));
  unsigned rem = gid - (unsigned)t * (N_NODES * FDIM);
  int n = (int)(rem >> 7);
  int f = (int)(rem & 127u);
  float cv = cnt[(size_t)t * N_NODES + n];
  float x  = s1[gid];
  float mean = (cv > 0.0f) ? x / fmaxf(cv, 1.0f) : 0.0f;
  float h = fmaxf(mean, 0.0f);
  int pf = t * FDIM + f;
  h = (h - m[pf]) * rsqrtf(v[pf] + BN_EPS) * g[pf] + b[pf];
  h1f[gid] = h;
  seq[((size_t)n * WIN + t) * (2 * FDIM) + f] = (__bf16)h;
  s1[gid] = 0.0f;                                   // ready for scatter #2
}

// relu -> BN on sum aggregate; bf16 into seq[:, t, F:2F]
__global__ void mpnn2_post(const float* __restrict__ s2,
                           const float* __restrict__ g, const float* __restrict__ b,
                           const float* __restrict__ m, const float* __restrict__ v,
                           __bf16* __restrict__ seq) {
  unsigned gid = blockIdx.x * blockDim.x + threadIdx.x;
  int t = (int)(gid / (N_NODES * FDIM));
  unsigned rem = gid - (unsigned)t * (N_NODES * FDIM);
  int n = (int)(rem >> 7);
  int f = (int)(rem & 127u);
  float h = fmaxf(s2[gid], 0.0f);
  int pf = t * FDIM + f;
  h = (h - m[pf]) * rsqrtf(v[pf] + BN_EPS) * g[pf] + b[pf];
  seq[((size_t)n * WIN + t) * (2 * FDIM) + FDIM + f] = (__bf16)h;
}

// ---------------------------------------------------------------------------
// Weight pre-pack: f32 [256,1024] row-major -> bf16 WMMA B-fragment layout
// Bp[((kt*64 + nt)*32 + lane)*16 + e] = B[kt*32 + (lane>>4)*16 + e][nt*16 + (lane&15)]
// so each lane's 16 bf16 fragment is two coalesced b128 loads in the GEMM.
// ---------------------------------------------------------------------------
__global__ void pack_b_kernel(const float* __restrict__ B, __bf16* __restrict__ Bp) {
  unsigned gid = blockIdx.x * blockDim.x + threadIdx.x;   // 8*64*32*16 = 262144
  int e  = (int)(gid & 15u);
  int l  = (int)((gid >> 4) & 31u);
  int nt = (int)((gid >> 9) & 63u);
  int kt = (int)(gid >> 15);
  int k = kt * 32 + ((l >> 4) << 4) + e;
  int c = nt * 16 + (l & 15);
  Bp[gid] = (__bf16)B[(size_t)k * GNCOLS + c];
}

// ---------------------------------------------------------------------------
// C[M,1024] = A[M,256](bf16) @ Bp(packed bf16) + (Cin | bias)
// Block = 8 waves; block covers 16 rows x 128 cols; wave = one 16x16 tile,
// 8 x v_wmma_f32_16x16x32_bf16 over K=256. A tile staged in LDS (8x reuse).
// ---------------------------------------------------------------------------
__global__ __launch_bounds__(256)
void gemm_wmma(const __bf16* __restrict__ A, int lda,
               const __bf16* __restrict__ Bp,
               const float* __restrict__ Cin,
               const float* __restrict__ bias,
               float* __restrict__ Cout, int M) {
  __shared__ __bf16 As[16][272];                 // 16-elem row pad (32B) vs bank conflicts
  const int tid  = threadIdx.x;
  const int wid  = tid >> 5;
  const int lane = tid & 31;
  const int m0   = blockIdx.x * 16;
  const int nt   = blockIdx.y * 8 + wid;         // 0..63
  const int col  = nt * 16 + (lane & 15);

  // cooperative stage of the 16x256 bf16 A tile (8 KB): 16 bf16 per thread
  {
    const int row = tid >> 4;
    const int cb  = (tid & 15) * 16;
    uint4 u0 = make_uint4(0, 0, 0, 0), u1 = make_uint4(0, 0, 0, 0);
    if (m0 + row < M) {
      const uint4* srcp = (const uint4*)(A + (size_t)(m0 + row) * lda + cb);
      u0 = srcp[0];
      u1 = srcp[1];
    }
    *(uint4*)&As[row][cb]     = u0;
    *(uint4*)&As[row][cb + 8] = u1;
  }
  __syncthreads();

  const int mrow = m0 + ((lane >> 4) << 3);      // C layout: v[r] = row mrow+r
  v8f acc;
  if (Cin != nullptr) {
#pragma unroll
    for (int r = 0; r < 8; ++r)
      acc[r] = (mrow + r < M) ? Cin[(size_t)(mrow + r) * GNCOLS + col] : 0.0f;
  } else {
    float bv = (bias != nullptr) ? bias[col] : 0.0f;
#pragma unroll
    for (int r = 0; r < 8; ++r) acc[r] = bv;
  }

  const int arow = lane & 15;                    // A frag: row = lane&15
  const int koff = (lane >> 4) << 4;             // lanes 16..31 hold K+16 slice
#pragma unroll
  for (int kt = 0; kt < 8; ++kt) {
    v8bf alo = *(const v8bf*)&As[arow][kt * 32 + koff];
    v8bf ahi = *(const v8bf*)&As[arow][kt * 32 + koff + 8];
    v16bf a = __builtin_shufflevector(alo, ahi, 0,1,2,3,4,5,6,7,8,9,10,11,12,13,14,15);

    const __bf16* bp = Bp + (((size_t)kt * 64 + nt) * 32 + lane) * 16;
    v8bf blo = ((const v8bf*)bp)[0];
    v8bf bhi = ((const v8bf*)bp)[1];
    v16bf b = __builtin_shufflevector(blo, bhi, 0,1,2,3,4,5,6,7,8,9,10,11,12,13,14,15);

    if (kt < 7) __builtin_prefetch(bp + (size_t)64 * 32 * 16, 0, 1);  // next K slab

    acc = __builtin_amdgcn_wmma_f32_16x16x32_bf16(
        /*neg_a=*/false, a, /*neg_b=*/false, b,
        /*c_mod=*/(short)0, acc, /*reuse_a=*/false, /*reuse_b=*/false);
  }

#pragma unroll
  for (int r = 0; r < 8; ++r)
    if (mrow + r < M) Cout[(size_t)(mrow + r) * GNCOLS + col] = acc[r];
}

// ---------------------------------------------------------------------------
// LSTM gates (keras order i,f,g,o), f32 cell state, bf16 hidden for next GEMM
// ---------------------------------------------------------------------------
__global__ void lstm_gate(const float* __restrict__ z, float* __restrict__ c,
                          __bf16* __restrict__ h, __bf16* __restrict__ hs, int t) {
  unsigned gid = blockIdx.x * blockDim.x + threadIdx.x;   // N*U threads
  int n = (int)(gid >> 8);
  int u = (int)(gid & 255u);
  size_t zb = (size_t)n * GNCOLS;
  float zi = z[zb + u];
  float zf = z[zb + UNITS + u];
  float zg = z[zb + 2 * UNITS + u];
  float zo = z[zb + 3 * UNITS + u];
  float si = 1.0f / (1.0f + __expf(-zi));
  float sf = 1.0f / (1.0f + __expf(-zf));
  float so = 1.0f / (1.0f + __expf(-zo));
  float cn = sf * c[gid] + si * tanhf(zg);
  float hn = so * tanhf(cn);
  c[gid] = cn;
  h[gid] = (__bf16)hn;
  if (hs != nullptr) hs[((size_t)n * WIN + t) * UNITS + u] = (__bf16)hn;
}

__global__ void dense_out(const __bf16* __restrict__ h,
                          const float* __restrict__ wd,
                          const float* __restrict__ bd,
                          float* __restrict__ out) {
  int n = (int)(blockIdx.x * blockDim.x + threadIdx.x);
  if (n >= N_NODES) return;
  float acc = bd[0];
  const __bf16* hp = h + (size_t)n * UNITS;
#pragma unroll 8
  for (int u = 0; u < UNITS; ++u) acc += (float)hp[u] * wd[u];
  out[n] = fmaxf(acc, 0.0f);
}

// ---------------------------------------------------------------------------
extern "C" void kernel_launch(void* const* d_in, const int* in_sizes, int n_in,
                              void* d_out, int out_size, void* d_ws, size_t ws_size,
                              hipStream_t stream) {
  const float* X      = (const float*)d_in[0];
  const int*   esrc   = (const int*)d_in[1];
  const int*   edst   = (const int*)d_in[2];
  const float* gamma1 = (const float*)d_in[3];
  const float* beta1  = (const float*)d_in[4];
  const float* mean1  = (const float*)d_in[5];
  const float* var1   = (const float*)d_in[6];
  const float* gamma2 = (const float*)d_in[7];
  const float* beta2  = (const float*)d_in[8];
  const float* mean2  = (const float*)d_in[9];
  const float* var2   = (const float*)d_in[10];
  const float* k1     = (const float*)d_in[11];
  const float* r1     = (const float*)d_in[12];
  const float* b1     = (const float*)d_in[13];
  const float* k2     = (const float*)d_in[14];
  const float* r2     = (const float*)d_in[15];
  const float* b2     = (const float*)d_in[16];
  const float* wd     = (const float*)d_in[17];
  const float* bd     = (const float*)d_in[18];

  // workspace carve (~345 MB total)
  char* w = (char*)d_ws;
  auto carve = [&](size_t bytes) -> char* {
    char* p = w;
    w += (bytes + 255) & ~(size_t)255;
    return p;
  };
  float*  s1  = (float*) carve((size_t)WIN * N_NODES * FDIM * 4);       // scatter accum (reused)
  float*  cnt = (float*) carve((size_t)WIN * N_NODES * 4);
  float*  h1f = (float*) carve((size_t)WIN * N_NODES * FDIM * 4);
  __bf16* seq = (__bf16*)carve((size_t)N_NODES * WIN * 2 * FDIM * 2);   // LSTM1 input, bf16
  __bf16* k1p = (__bf16*)carve((size_t)262144 * 2);
  __bf16* r1p = (__bf16*)carve((size_t)262144 * 2);
  __bf16* k2p = (__bf16*)carve((size_t)262144 * 2);
  __bf16* r2p = (__bf16*)carve((size_t)262144 * 2);
  float*  z   = (float*) carve((size_t)N_NODES * GNCOLS * 4);           // gate pre-activations
  float*  cst = (float*) carve((size_t)N_NODES * UNITS * 4);            // cell state
  __bf16* hbf = (__bf16*)carve((size_t)N_NODES * UNITS * 2);            // hidden (bf16)
  __bf16* hs  = (__bf16*)carve((size_t)N_NODES * WIN * UNITS * 2);      // LSTM1 outputs

  hipMemsetAsync(s1, 0, (size_t)WIN * N_NODES * FDIM * 4, stream);
  hipMemsetAsync(cnt, 0, (size_t)WIN * N_NODES * 4, stream);

  // weight packing (once per launch; deterministic)
  pack_b_kernel<<<1024, 256, 0, stream>>>(k1, k1p);
  pack_b_kernel<<<1024, 256, 0, stream>>>(r1, r1p);
  pack_b_kernel<<<1024, 256, 0, stream>>>(k2, k2p);
  pack_b_kernel<<<1024, 256, 0, stream>>>(r2, r2p);

  // graph phase (all 6 timesteps in one grid)
  const unsigned sc_threads = (unsigned)WIN * EDGES * 32u;              // 122,880,000
  const unsigned sc_blocks  = (sc_threads + 255u) / 256u;
  const unsigned el_threads = (unsigned)WIN * N_NODES * FDIM;           // 15,360,000
  const unsigned el_blocks  = el_threads / 256u;

  scatter_kernel<<<sc_blocks, 256, 0, stream>>>(X, esrc, edst, s1, cnt, sc_threads);
  mpnn1_post<<<el_blocks, 256, 0, stream>>>(s1, cnt, gamma1, beta1, mean1, var1, h1f, seq);
  scatter_kernel<<<sc_blocks, 256, 0, stream>>>(h1f, esrc, edst, s1, nullptr, sc_threads);
  mpnn2_post<<<el_blocks, 256, 0, stream>>>(s1, gamma2, beta2, mean2, var2, seq);

  const dim3 ggrid((N_NODES + 15) / 16, 8);
  const unsigned gt_blocks = ((unsigned)N_NODES * UNITS) / 256u;

  // LSTM 1
  hipMemsetAsync(hbf, 0, (size_t)N_NODES * UNITS * 2, stream);
  hipMemsetAsync(cst, 0, (size_t)N_NODES * UNITS * 4, stream);
  for (int t = 0; t < WIN; ++t) {
    gemm_wmma<<<ggrid, 256, 0, stream>>>(seq + (size_t)t * 2 * FDIM, WIN * 2 * FDIM,
                                         k1p, nullptr, b1, z, N_NODES);
    gemm_wmma<<<ggrid, 256, 0, stream>>>(hbf, UNITS, r1p, z, nullptr, z, N_NODES);
    lstm_gate<<<gt_blocks, 256, 0, stream>>>(z, cst, hbf, hs, t);
  }

  // LSTM 2
  hipMemsetAsync(hbf, 0, (size_t)N_NODES * UNITS * 2, stream);
  hipMemsetAsync(cst, 0, (size_t)N_NODES * UNITS * 4, stream);
  for (int t = 0; t < WIN; ++t) {
    gemm_wmma<<<ggrid, 256, 0, stream>>>(hs + (size_t)t * UNITS, WIN * UNITS,
                                         k2p, nullptr, b2, z, N_NODES);
    gemm_wmma<<<ggrid, 256, 0, stream>>>(hbf, UNITS, r2p, z, nullptr, z, N_NODES);
    lstm_gate<<<gt_blocks, 256, 0, stream>>>(z, cst, hbf, nullptr, 0);
  }

  dense_out<<<(N_NODES + 255) / 256, 256, 0, stream>>>(hbf, wd, bd, (float*)d_out);
}
